// RealSpaceEGNNEncoder_5806795784728
// MI455X (gfx1250) — compile-verified, matrix operand order checked by reading, not run on hardware
//
#include <hip/hip_runtime.h>
#include <math.h>

typedef __attribute__((ext_vector_type(16))) _Float16 v16h;
typedef __attribute__((ext_vector_type(8)))  _Float16 v8h;
typedef __attribute__((ext_vector_type(8)))  float    v8f;

#define NSC 64
#define NVC 32
#define NTC 16
#define NGR 64
#define LATENT 128
#define N_LAYERS 4

// f16 weight block offsets (halves) within one layer, canonical order
#define OFF_MS   0
#define OFF_MV   6144
#define OFF_MT   10240
#define OFF_MGV  11264
#define OFF_MGT  13312
#define OFF_MLS  14336
#define OFF_MLV  18432
#define OFF_MLT  19456
#define OFF_US   19968
#define OFF_UV   28160
#define OFF_UT   31232
#define OFF_UGV  32768
#define OFF_UGT  34816
#define OFF_ULS  35840
#define OFF_ULV  39936
#define OFF_ULT  40960
#define LAYER_HALVES 41472

// ---------------- WMMA fragment helpers (CDNA5 16x16x32 f16 layouts) ----------------

__device__ __forceinline__ v16h load_fragA(const _Float16* base, int stride, int k0) {
  int lane = threadIdx.x & 31;
  const _Float16* p = base + (lane & 15) * stride + k0 + ((lane >> 4) & 1) * 8;
  v8h a0 = *(const v8h*)(p);
  v8h a1 = *(const v8h*)(p + 16);
  v16h r;
#pragma unroll
  for (int i = 0; i < 8; ++i) { r[i] = a0[i]; r[i + 8] = a1[i]; }
  return r;
}

__device__ __forceinline__ v16h load_fragB(const _Float16* Wt, int Kp, int k0, int n0) {
  int lane = threadIdx.x & 31;
  const _Float16* p = Wt + (size_t)((lane & 15) + n0) * Kp + k0 + ((lane >> 4) & 1) * 8;
  v8h b0 = *(const v8h*)(p);
  v8h b1 = *(const v8h*)(p + 16);
  v16h r;
#pragma unroll
  for (int i = 0; i < 8; ++i) { r[i] = b0[i]; r[i + 8] = b1[i]; }
  return r;
}

// D(16x16) += A(16xK) * B(KxN-tile) ; A in LDS row-major (stride halves), B = f16 weights [out][Kp]
__device__ __forceinline__ v8f mmacc(const _Float16* A, int As, const _Float16* B, int Bp,
                                     int n0, int ksteps) {
  v8f acc = {0.f,0.f,0.f,0.f,0.f,0.f,0.f,0.f};
  for (int k = 0; k < ksteps; ++k) {
    v16h a = load_fragA(A, As, k * 32);
    v16h b = load_fragB(B, Bp, k * 32, n0);
    acc = __builtin_amdgcn_wmma_f32_16x16x32_f16(false, a, false, b, (short)0, acc, false, false);
  }
  return acc;
}

// store C fragment as f16 into row-major [16][stride] LDS (A-layout for the next matmul)
__device__ __forceinline__ void storeC_h(_Float16* out, int stride, int n0, v8f c, bool dosilu) {
  int lane = threadIdx.x & 31;
  int col = (lane & 15) + n0;
  int hi = (lane >> 4) & 1;
#pragma unroll
  for (int r = 0; r < 8; ++r) {
    float x = c[r];
    if (dosilu) x = x / (1.f + expf(-x));
    out[(r + hi * 8) * stride + col] = (_Float16)x;
  }
}

__device__ __forceinline__ v8f sigm8(v8f x) {
  v8f r;
#pragma unroll
  for (int i = 0; i < 8; ++i) r[i] = 1.f / (1.f + expf(-x[i]));
  return r;
}

// scatter-add C fragment into per-node arrays via LDS node index list
__device__ __forceinline__ void scatterC(float* base, const int* nodes, int nodeStride,
                                         int chStride, int compOff, int n0, v8f c) {
  int lane = threadIdx.x & 31;
  int col = (lane & 15) + n0;
  int hi = (lane >> 4) & 1;
#pragma unroll
  for (int r = 0; r < 8; ++r) {
    int m = r + (hi << 3);
    atomicAdd(base + (size_t)nodes[m] * nodeStride + (size_t)col * chStride + compOff, c[r]);
  }
}

// dense residual add (node = n0node + m)
__device__ __forceinline__ void addC_dense(float* base, int n0node, int N, int nodeStride,
                                           int chStride, int compOff, int n0, v8f c) {
  int lane = threadIdx.x & 31;
  int col = (lane & 15) + n0;
  int hi = (lane >> 4) & 1;
#pragma unroll
  for (int r = 0; r < 8; ++r) {
    int nd = n0node + r + (hi << 3);
    if (nd < N)
      atomicAdd(base + (size_t)nd * nodeStride + (size_t)col * chStride + compOff, c[r]);
  }
}

// ---------------- small utility kernels ----------------

__global__ void wprep_kernel(const float* __restrict__ src, _Float16* __restrict__ dst,
                             int K, int O, int Kp, int n) {
  int i = blockIdx.x * blockDim.x + threadIdx.x;
  if (i >= n) return;
  int o = i / Kp, k = i - o * Kp;
  dst[i] = (k < K) ? (_Float16)src[(size_t)k * O + o] : (_Float16)0.f;
}

__global__ void zero_kernel(float* p, long long n) {
  long long i = (long long)blockIdx.x * blockDim.x + threadIdx.x;
  if (i < n) p[i] = 0.f;
}

__global__ void init_s_kernel(const float* __restrict__ x, const float* __restrict__ W0,
                              float* __restrict__ S, int N) {
  int i = blockIdx.x * blockDim.x + threadIdx.x;
  if (i >= N * 64) return;
  int n = i >> 6, c = i & 63;
  S[i] = x[n * 3 + 0] * W0[c] + x[n * 3 + 1] * W0[64 + c] + x[n * 3 + 2] * W0[128 + c];
}

__global__ void edge_prep_kernel(const float* __restrict__ pos, const int* __restrict__ erow,
                                 const int* __restrict__ ecol, float* __restrict__ es,
                                 float* __restrict__ ev, int E) {
  int e = blockIdx.x * blockDim.x + threadIdx.x;
  if (e >= E) return;
  int a = erow[e], b = ecol[e];
  float rx = pos[a * 3 + 0] - pos[b * 3 + 0];
  float ry = pos[a * 3 + 1] - pos[b * 3 + 1];
  float rz = pos[a * 3 + 2] - pos[b * 3 + 2];
  float d = sqrtf(rx * rx + ry * ry + rz * rz);
  es[e] = d;
  float inv = 1.f / (d + 1e-8f);
  ev[e * 3 + 0] = rx * inv;
  ev[e * 3 + 1] = ry * inv;
  ev[e * 3 + 2] = rz * inv;
}

__global__ void pool_accum_kernel(const float* __restrict__ S, const int* __restrict__ batch,
                                  float* __restrict__ pooled, float* __restrict__ cnt, int N) {
  int i = blockIdx.x * blockDim.x + threadIdx.x;
  if (i >= N * 64) return;
  int n = i >> 6, c = i & 63;
  int g = batch[n];
  atomicAdd(&pooled[g * 64 + c], S[i]);
  if (c == 0) atomicAdd(&cnt[g], 1.f);
}

__global__ void final_mm_kernel(const float* __restrict__ pooled, const float* __restrict__ cnt,
                                const float* __restrict__ Wf, float* __restrict__ out) {
  int i = blockIdx.x * blockDim.x + threadIdx.x;
  if (i >= NGR * LATENT) return;
  int g = i / LATENT, l = i - g * LATENT;
  float inv = 1.f / fmaxf(cnt[g], 1.f);
  float acc = 0.f;
  for (int c = 0; c < 64; ++c) acc += pooled[g * 64 + c] * inv * Wf[c * LATENT + l];
  out[i] = acc;
}

// ---------------- message kernel: one wave = 16 edges ----------------

#define MSG_WAVES 2

__global__ void __launch_bounds__(64) msg_kernel(
    const float* __restrict__ S, const float* __restrict__ V, const float* __restrict__ T,
    const int* __restrict__ erow, const int* __restrict__ ecol,
    const float* __restrict__ es, const float* __restrict__ ev,
    const _Float16* __restrict__ W,
    float* __restrict__ AS, float* __restrict__ AV, float* __restrict__ AT, int E) {
  __shared__ __align__(16) _Float16 Ab[MSG_WAVES][16 * 128];
  __shared__ __align__(16) _Float16 Sb[MSG_WAVES][16 * 64];
  __shared__ float Sg_[MSG_WAVES][16 * 64];
  __shared__ float Vg_[MSG_WAVES][16 * 96];
  __shared__ float Tg_[MSG_WAVES][16 * 144];
  __shared__ int nid[MSG_WAVES][16];
  __shared__ int srcb[MSG_WAVES][16];
  __shared__ float esb[MSG_WAVES][16];
  __shared__ float evb[MSG_WAVES][16][3];

  const int wid = threadIdx.x >> 5;
  const int lane = threadIdx.x & 31;
  const int e0 = (blockIdx.x * MSG_WAVES + wid) * 16;
  _Float16* A = Ab[wid];
  _Float16* Sm = Sb[wid];
  float* sg = Sg_[wid];
  float* vg = Vg_[wid];
  float* tg = Tg_[wid];

  if (lane < 16) {
    int e = e0 + lane;
    bool ok = e < E;
    nid[wid][lane] = ok ? erow[e] : 0;
    srcb[wid][lane] = ok ? ecol[e] : 0;
    esb[wid][lane] = ok ? es[e] : 0.f;
    evb[wid][lane][0] = ok ? ev[3 * e + 0] : 0.f;
    evb[wid][lane][1] = ok ? ev[3 * e + 1] : 0.f;
    evb[wid][lane][2] = ok ? ev[3 * e + 2] : 0.f;
  }
  __builtin_amdgcn_wave_barrier();

  // stage source-node features into LDS once
  for (int idx = lane; idx < 16 * 64; idx += 32) { int m = idx >> 6; sg[idx] = S[(size_t)srcb[wid][m] * 64 + (idx & 63)]; }
  for (int idx = lane; idx < 16 * 96; idx += 32) { int m = idx / 96; vg[idx] = V[(size_t)srcb[wid][m] * 96 + (idx - m * 96)]; }
  for (int idx = lane; idx < 16 * 144; idx += 32) { int m = idx / 144; tg[idx] = T[(size_t)srcb[wid][m] * 144 + (idx - m * 144)]; }
  __builtin_amdgcn_wave_barrier();

  // ---- ps[16][96] -> ms = ps @ Wms ----
  for (int idx = lane; idx < 16 * 96; idx += 32) {
    int m = idx / 96, c = idx - m * 96;
    float val;
    if (c < 64) val = sg[m * 64 + c] * esb[wid][m];
    else {
      int cc = c - 64;
      const float* vv = &vg[m * 96 + cc * 3];
      val = vv[0] * evb[wid][m][0] + vv[1] * evb[wid][m][1] + vv[2] * evb[wid][m][2];
    }
    A[m * 128 + c] = (_Float16)val;
  }
  __builtin_amdgcn_wave_barrier();

  v8f accS[4];
#pragma unroll
  for (int nt = 0; nt < 4; ++nt) accS[nt] = mmacc(A, 128, W + OFF_MS, 96, nt * 16, 3);
#pragma unroll
  for (int nt = 0; nt < 4; ++nt) storeC_h(Sm, 64, nt * 16, accS[nt], false);  // raw ms
  __builtin_amdgcn_wave_barrier();

  v8f gv0 = sigm8(mmacc(Sm, 64, W + OFF_MGV, 64, 0, 2));
  v8f gv1 = sigm8(mmacc(Sm, 64, W + OFF_MGV, 64, 16, 2));
  v8f gt0 = sigm8(mmacc(Sm, 64, W + OFF_MGT, 64, 0, 2));

#pragma unroll
  for (int nt = 0; nt < 4; ++nt) storeC_h(Sm, 64, nt * 16, accS[nt], true);  // silu(ms)
  __builtin_amdgcn_wave_barrier();
#pragma unroll
  for (int nt = 0; nt < 4; ++nt) {
    v8f o = mmacc(Sm, 64, W + OFF_MLS, 64, nt * 16, 2);
    scatterC(AS, nid[wid], 64, 1, 0, nt * 16, o);
  }

  // ---- pv[16][112] per vector component -> mv ----
  for (int i = 0; i < 3; ++i) {
    for (int idx = lane; idx < 16 * 128; idx += 32) {
      int m = idx >> 7, c = idx & 127;
      float val = 0.f;
      if (c < 64) val = sg[m * 64 + c] * evb[wid][m][i];
      else if (c < 96) { int cc = c - 64; val = vg[m * 96 + cc * 3 + i] * esb[wid][m]; }
      else if (c < 112) {
        int cc = c - 96;
        const float* tt = &tg[m * 144 + cc * 9 + i * 3];
        val = tt[0] * evb[wid][m][0] + tt[1] * evb[wid][m][1] + tt[2] * evb[wid][m][2];
      }
      A[m * 128 + c] = (_Float16)val;
    }
    __builtin_amdgcn_wave_barrier();
    v8f f0 = mmacc(A, 128, W + OFF_MV, 128, 0, 4) * gv0;
    v8f f1 = mmacc(A, 128, W + OFF_MV, 128, 16, 4) * gv1;
    __builtin_amdgcn_wave_barrier();
    storeC_h(A, 128, 0, f0, false);
    storeC_h(A, 128, 16, f1, false);
    __builtin_amdgcn_wave_barrier();
#pragma unroll
    for (int nt = 0; nt < 2; ++nt) {
      v8f o = mmacc(A, 128, W + OFF_MLV, 32, nt * 16, 1);
      scatterC(AV, nid[wid], 96, 3, i, nt * 16, o);
    }
  }

  // ---- pt[16][48] per tensor component -> mt ----
  for (int q = 0; q < 9; ++q) {
    int i = q / 3, j = q - i * 3;
    for (int idx = lane; idx < 16 * 64; idx += 32) {
      int m = idx >> 6, c = idx & 63;
      float val = 0.f;
      if (c < 16) val = tg[m * 144 + c * 9 + q] * esb[wid][m];
      else if (c < 48) {
        int cc = c - 16;
        const float* vv = &vg[m * 96 + cc * 3];
        float evi = evb[wid][m][i], evj = evb[wid][m][j];
        val = 0.5f * (vv[i] * evj + vv[j] * evi);
        if (i == j)
          val -= (vv[0] * evb[wid][m][0] + vv[1] * evb[wid][m][1] + vv[2] * evb[wid][m][2]) * (1.f / 3.f);
      }
      A[m * 128 + c] = (_Float16)val;
    }
    __builtin_amdgcn_wave_barrier();
    v8f f = mmacc(A, 128, W + OFF_MT, 64, 0, 2) * gt0;
    __builtin_amdgcn_wave_barrier();
    storeC_h(A, 128, 0, f, false);
    for (int idx = lane; idx < 16 * 16; idx += 32) { int m = idx >> 4; A[m * 128 + 16 + (idx & 15)] = (_Float16)0.f; }
    __builtin_amdgcn_wave_barrier();
    v8f o = mmacc(A, 128, W + OFF_MLT, 32, 0, 1);
    scatterC(AT, nid[wid], 144, 9, q, 0, o);
  }
}

// ---------------- update kernel: one wave = 16 nodes ----------------

#define UPD_WAVES 2

__global__ void __launch_bounds__(64) upd_kernel(
    float* __restrict__ S, float* __restrict__ V, float* __restrict__ T,
    const float* __restrict__ AS, const float* __restrict__ AV, const float* __restrict__ AT,
    const _Float16* __restrict__ W, int N) {
  __shared__ __align__(16) _Float16 Ab[UPD_WAVES][16 * 128];
  __shared__ __align__(16) _Float16 Sb[UPD_WAVES][16 * 64];
  __shared__ float Sg_[UPD_WAVES][16 * 64];
  __shared__ float Vg_[UPD_WAVES][16 * 96];
  __shared__ float Tg_[UPD_WAVES][16 * 144];

  const int wid = threadIdx.x >> 5;
  const int lane = threadIdx.x & 31;
  const int n0 = (blockIdx.x * UPD_WAVES + wid) * 16;
  _Float16* A = Ab[wid];
  _Float16* Sm = Sb[wid];
  float* sg = Sg_[wid];
  float* vg = Vg_[wid];
  float* tg = Tg_[wid];

  for (int idx = lane; idx < 16 * 64; idx += 32) { int m = idx >> 6; int nd = n0 + m; sg[idx] = (nd < N) ? S[(size_t)nd * 64 + (idx & 63)] : 0.f; }
  for (int idx = lane; idx < 16 * 96; idx += 32) { int m = idx / 96; int nd = n0 + m; vg[idx] = (nd < N) ? V[(size_t)nd * 96 + (idx - m * 96)] : 0.f; }
  for (int idx = lane; idx < 16 * 144; idx += 32) { int m = idx / 144; int nd = n0 + m; tg[idx] = (nd < N) ? T[(size_t)nd * 144 + (idx - m * 144)] : 0.f; }
  __builtin_amdgcn_wave_barrier();

  // ---- us input [16][112] ----
  for (int idx = lane; idx < 16 * 128; idx += 32) {
    int m = idx >> 7, c = idx & 127;
    int nd = n0 + m;
    float val = 0.f;
    if (nd < N) {
      if (c < 64) val = sg[m * 64 + c] * AS[(size_t)nd * 64 + c];
      else if (c < 96) {
        int cc = c - 64;
        const float* av = &AV[(size_t)nd * 96 + cc * 3];
        const float* vv = &vg[m * 96 + cc * 3];
        val = vv[0] * av[0] + vv[1] * av[1] + vv[2] * av[2];
      } else if (c < 112) {
        int cc = c - 96;
        const float* at = &AT[(size_t)nd * 144 + cc * 9];
        const float* tt = &tg[m * 144 + cc * 9];
        float a = 0.f;
#pragma unroll
        for (int k = 0; k < 9; ++k) a += tt[k] * at[k];
        val = a;
      }
    }
    A[m * 128 + c] = (_Float16)val;
  }
  __builtin_amdgcn_wave_barrier();

  v8f accU[4];
#pragma unroll
  for (int nt = 0; nt < 4; ++nt) accU[nt] = mmacc(A, 128, W + OFF_US, 128, nt * 16, 4);
#pragma unroll
  for (int nt = 0; nt < 4; ++nt) storeC_h(Sm, 64, nt * 16, accU[nt], false);
  __builtin_amdgcn_wave_barrier();

  v8f gv0 = sigm8(mmacc(Sm, 64, W + OFF_UGV, 64, 0, 2));
  v8f gv1 = sigm8(mmacc(Sm, 64, W + OFF_UGV, 64, 16, 2));
  v8f gt0 = sigm8(mmacc(Sm, 64, W + OFF_UGT, 64, 0, 2));

#pragma unroll
  for (int nt = 0; nt < 4; ++nt) storeC_h(Sm, 64, nt * 16, accU[nt], true);
  __builtin_amdgcn_wave_barrier();
#pragma unroll
  for (int nt = 0; nt < 4; ++nt) {
    v8f o = mmacc(Sm, 64, W + OFF_ULS, 64, nt * 16, 2);
    addC_dense(S, n0, N, 64, 1, 0, nt * 16, o);
  }

  // ---- upv [16][96] per component ----
  for (int i = 0; i < 3; ++i) {
    for (int idx = lane; idx < 16 * 96; idx += 32) {
      int m = idx / 96, c = idx - m * 96;
      int nd = n0 + m;
      float val = 0.f;
      if (nd < N) {
        if (c < 32) val = sg[m * 64 + c] * AV[(size_t)nd * 96 + c * 3 + i];
        else if (c < 64) { int cc = c - 32; val = vg[m * 96 + cc * 3 + i] * AS[(size_t)nd * 64 + cc]; }
        else if (c < 80) {
          int cc = c - 64;
          const float* tt = &tg[m * 144 + cc * 9 + i * 3];
          const float* av = &AV[(size_t)nd * 96 + cc * 3];
          val = tt[0] * av[0] + tt[1] * av[1] + tt[2] * av[2];
        } else {
          int cc = c - 80;
          const float* at = &AT[(size_t)nd * 144 + cc * 9 + i * 3];
          const float* vv = &vg[m * 96 + cc * 3];
          val = at[0] * vv[0] + at[1] * vv[1] + at[2] * vv[2];
        }
      }
      A[m * 128 + c] = (_Float16)val;
    }
    __builtin_amdgcn_wave_barrier();
    v8f f0 = mmacc(A, 128, W + OFF_UV, 96, 0, 3) * gv0;
    v8f f1 = mmacc(A, 128, W + OFF_UV, 96, 16, 3) * gv1;
    __builtin_amdgcn_wave_barrier();
    storeC_h(A, 128, 0, f0, false);
    storeC_h(A, 128, 16, f1, false);
    __builtin_amdgcn_wave_barrier();
#pragma unroll
    for (int nt = 0; nt < 2; ++nt) {
      v8f o = mmacc(A, 128, W + OFF_ULV, 32, nt * 16, 1);
      addC_dense(V, n0, N, 96, 3, i, nt * 16, o);
    }
  }

  // ---- upt [16][80] per component ----
  for (int q = 0; q < 9; ++q) {
    int i = q / 3, j = q - i * 3;
    for (int idx = lane; idx < 16 * 96; idx += 32) {
      int m = idx / 96, c = idx - m * 96;
      int nd = n0 + m;
      float val = 0.f;
      if (nd < N) {
        if (c < 16) val = tg[m * 144 + c * 9 + q] * AS[(size_t)nd * 64 + c];
        else if (c < 32) { int cc = c - 16; val = AT[(size_t)nd * 144 + cc * 9 + q] * sg[m * 64 + cc]; }
        else if (c < 64) {
          int cc = c - 32;
          const float* vv = &vg[m * 96 + cc * 3];
          const float* av = &AV[(size_t)nd * 96 + cc * 3];
          val = 0.5f * (vv[i] * av[j] + vv[j] * av[i]);
          if (i == j) val -= (vv[0] * av[0] + vv[1] * av[1] + vv[2] * av[2]) * (1.f / 3.f);
        } else if (c < 80) {
          int cc = c - 64;
          const float* tt = &tg[m * 144 + cc * 9];
          const float* at = &AT[(size_t)nd * 144 + cc * 9];
          float pij = 0.f, pji = 0.f;
#pragma unroll
          for (int k = 0; k < 3; ++k) { pij += tt[i * 3 + k] * at[k * 3 + j]; pji += tt[j * 3 + k] * at[k * 3 + i]; }
          val = 0.5f * (pij + pji);
          if (i == j) {
            float tr = 0.f;
#pragma unroll
            for (int a2 = 0; a2 < 3; ++a2)
#pragma unroll
              for (int k = 0; k < 3; ++k) tr += tt[a2 * 3 + k] * at[k * 3 + a2];
            val -= tr * (1.f / 3.f);
          }
        }
      }
      A[m * 128 + c] = (_Float16)val;
    }
    __builtin_amdgcn_wave_barrier();
    v8f f = mmacc(A, 128, W + OFF_UT, 96, 0, 3) * gt0;
    __builtin_amdgcn_wave_barrier();
    storeC_h(A, 128, 0, f, false);
    for (int idx = lane; idx < 16 * 16; idx += 32) { int m = idx >> 4; A[m * 128 + 16 + (idx & 15)] = (_Float16)0.f; }
    __builtin_amdgcn_wave_barrier();
    v8f o = mmacc(A, 128, W + OFF_ULT, 32, 0, 1);
    addC_dense(T, n0, N, 144, 9, q, 0, o);
  }
}

// ---------------- host ----------------

extern "C" void kernel_launch(void* const* d_in, const int* in_sizes, int n_in,
                              void* d_out, int out_size, void* d_ws, size_t ws_size,
                              hipStream_t stream) {
  (void)out_size; (void)ws_size;
  // canonical per-layer weight order: ms,mv,mt,mgv,mgt,mls,mlv,mlt,us,uv,ut,ugv,ugt,uls,ulv,ult
  static const int mapA[16] = {0,1,2,3,4,5,6,7,8,9,10,11,12,13,14,15};           // insertion order
  static const int mapB[16] = {5,7,6,1,0,2,4,3,13,15,14,9,8,10,12,11};           // sorted-key pytree order
  int ix, ipos, iedge, ibatch, iW0, iWf, lbase;
  const int* lmap;
  if (n_in >= 2 && in_sizes[0] == in_sizes[1]) {        // x(3N), pos(3N) first => insertion order
    ix = 0; ipos = 1; iedge = 2; ibatch = 3; iW0 = 4; iWf = 5; lbase = 6; lmap = mapA;
  } else {                                              // batch, edge_index, W0, Wf, layers..., pos, x
    ibatch = 0; iedge = 1; iW0 = 2; iWf = 3; lbase = 4; lmap = mapB; ipos = n_in - 2; ix = n_in - 1;
  }
  const int N = in_sizes[ibatch];
  const int E = in_sizes[iedge] / 2;

  const float* x = (const float*)d_in[ix];
  const float* pos = (const float*)d_in[ipos];
  const int* ei = (const int*)d_in[iedge];
  const int* batch = (const int*)d_in[ibatch];
  const float* W0 = (const float*)d_in[iW0];
  const float* Wf = (const float*)d_in[iWf];
  const int* erow = ei;
  const int* ecol = ei + E;

  char* base = (char*)d_ws;
  size_t off = 0;
  auto alloc = [&](size_t bytes) -> void* {
    void* p = base + off;
    off = (off + bytes + 255) & ~(size_t)255;
    return p;
  };
  float* S = (float*)alloc((size_t)N * 64 * 4);
  float* V = (float*)alloc((size_t)N * 96 * 4);
  float* T = (float*)alloc((size_t)N * 144 * 4);
  float* AS = (float*)alloc((size_t)N * 64 * 4);
  float* AV = (float*)alloc((size_t)N * 96 * 4);
  float* AT = (float*)alloc((size_t)N * 144 * 4);
  float* es = (float*)alloc((size_t)E * 4);
  float* ev = (float*)alloc((size_t)E * 12);
  float* pooled = (float*)alloc((size_t)NGR * NSC * 4);
  float* cnt = (float*)alloc((size_t)NGR * 4);
  _Float16* wf16 = (_Float16*)alloc((size_t)N_LAYERS * LAYER_HALVES * 2);

  struct WI { int K, O, Kp, woff; };
  static const WI wi[16] = {
      {96, 64, 96, OFF_MS},   {112, 32, 128, OFF_MV}, {48, 16, 64, OFF_MT},
      {64, 32, 64, OFF_MGV},  {64, 16, 64, OFF_MGT},  {64, 64, 64, OFF_MLS},
      {32, 32, 32, OFF_MLV},  {16, 16, 32, OFF_MLT},  {112, 64, 128, OFF_US},
      {96, 32, 96, OFF_UV},   {80, 16, 96, OFF_UT},   {64, 32, 64, OFF_UGV},
      {64, 16, 64, OFF_UGT},  {64, 64, 64, OFF_ULS},  {32, 32, 32, OFF_ULV},
      {16, 16, 32, OFF_ULT}};

  for (int l = 0; l < N_LAYERS; ++l)
    for (int m = 0; m < 16; ++m) {
      const float* src = (const float*)d_in[lbase + l * 16 + lmap[m]];
      _Float16* dst = wf16 + (size_t)l * LAYER_HALVES + wi[m].woff;
      int n = wi[m].O * wi[m].Kp;
      wprep_kernel<<<(n + 255) / 256, 256, 0, stream>>>(src, dst, wi[m].K, wi[m].O, wi[m].Kp, n);
    }

  init_s_kernel<<<(N * 64 + 255) / 256, 256, 0, stream>>>(x, W0, S, N);
  zero_kernel<<<((long long)N * 96 + 255) / 256, 256, 0, stream>>>(V, (long long)N * 96);
  zero_kernel<<<((long long)N * 144 + 255) / 256, 256, 0, stream>>>(T, (long long)N * 144);
  edge_prep_kernel<<<(E + 255) / 256, 256, 0, stream>>>(pos, erow, ecol, es, ev, E);

  const int etiles = (E + 15) / 16;
  const int ntiles = (N + 15) / 16;
  for (int l = 0; l < N_LAYERS; ++l) {
    const _Float16* Wl = wf16 + (size_t)l * LAYER_HALVES;
    zero_kernel<<<((long long)N * 64 + 255) / 256, 256, 0, stream>>>(AS, (long long)N * 64);
    zero_kernel<<<((long long)N * 96 + 255) / 256, 256, 0, stream>>>(AV, (long long)N * 96);
    zero_kernel<<<((long long)N * 144 + 255) / 256, 256, 0, stream>>>(AT, (long long)N * 144);
    msg_kernel<<<(etiles + MSG_WAVES - 1) / MSG_WAVES, MSG_WAVES * 32, 0, stream>>>(
        S, V, T, erow, ecol, es, ev, Wl, AS, AV, AT, E);
    upd_kernel<<<(ntiles + UPD_WAVES - 1) / UPD_WAVES, UPD_WAVES * 32, 0, stream>>>(
        S, V, T, AS, AV, AT, Wl, N);
  }

  zero_kernel<<<(NGR * NSC + 255) / 256, 256, 0, stream>>>(pooled, NGR * NSC);
  zero_kernel<<<(NGR + 255) / 256, 256, 0, stream>>>(cnt, NGR);
  pool_accum_kernel<<<(N * 64 + 255) / 256, 256, 0, stream>>>(S, batch, pooled, cnt, N);
  final_mm_kernel<<<(NGR * LATENT + 255) / 256, 256, 0, stream>>>(pooled, cnt, Wf, (float*)d_out);
}
